// My_RNN_Layer_68324339745003
// MI455X (gfx1250) — matrix operand order repaired and checked by reading, not compiled, not run on hardware
//
#include <hip/hip_runtime.h>
#include <hip/hip_bf16.h>

#define BATCH 128
#define TLEN  512
#define VOCAB 1024
#define HID   256

typedef __attribute__((ext_vector_type(16))) __bf16 v16bf;
typedef __attribute__((ext_vector_type(8)))  float  v8f;

union FragBF {
    uint4 u[2];
    v16bf v;
};

union Pack8BF {
    uint4  u;
    __bf16 h[8];
};

union Pack4BF {
    uint2  u;
    __bf16 h[4];
};

// Hardware tanh (V_TANH_F32) if the toolchain exposes it; else libm fallback.
#if defined(__has_builtin)
#if __has_builtin(__builtin_amdgcn_tanhf)
#define FAST_TANH(x) __builtin_amdgcn_tanhf(x)
#endif
#endif
#ifndef FAST_TANH
#define FAST_TANH(x) tanhf(x)
#endif

// Scheduling fence: keep the ds_load block and the WMMA chain separate so the
// A/B fragments stay in distinct registers and DScnt waits pipeline.
#if defined(__has_builtin)
#if __has_builtin(__builtin_amdgcn_sched_barrier)
#define SCHED_FENCE() __builtin_amdgcn_sched_barrier(0)
#endif
#endif
#ifndef SCHED_FENCE
#define SCHED_FENCE()
#endif

// ---------------------------------------------------------------------------
// Kernel 1: convert Wd (f32 [HID,VOCAB]) to bf16 in workspace (vectorized)
// ---------------------------------------------------------------------------
__global__ void wd_to_bf16(const float4* __restrict__ wd, uint2* __restrict__ wdb, int n4) {
    int i = blockIdx.x * blockDim.x + threadIdx.x;
    if (i < n4) {
        float4 f = wd[i];
        Pack4BF p;
        p.h[0] = (__bf16)f.x; p.h[1] = (__bf16)f.y;
        p.h[2] = (__bf16)f.z; p.h[3] = (__bf16)f.w;
        wdb[i] = p.u;
    }
}

// ---------------------------------------------------------------------------
// Kernel 2: sequential RNN scan, single workgroup (16 waves), WMMA bf16.
//   h_{t+1} = tanh(Wxh[idx[:,t]] + bh + h_t @ Whh)
// Each wave owns 16 HID columns; all Whh B-fragments live in VGPRs.
// ---------------------------------------------------------------------------
__global__ __launch_bounds__(512)
void rnn_scan(const int*   __restrict__ inp,    // [BATCH, TLEN]
              const float* __restrict__ state,  // [BATCH, HID]
              const float* __restrict__ Wxh,    // [VOCAB, HID]
              const float* __restrict__ Whh,    // [HID, HID]
              const float* __restrict__ bh,     // [HID]
              __bf16*      __restrict__ Ybf,    // [TLEN*BATCH, HID]
              float*       __restrict__ hfin)   // [BATCH, HID]
{
    __shared__ __bf16 sH[2][BATCH * HID];   // 2 x 64 KB double-buffered state
    __shared__ int    sRow[2][BATCH];       // gathered vocab row index per batch

    const int tid  = threadIdx.x;
    const int wave = tid >> 5;
    const int lane = tid & 31;
    const int l15  = lane & 15;
    const int sel  = lane >> 4;          // half-wave select
    const int nb   = wave * 16;          // this wave's N-tile base (HID cols)
    const int ncol = nb + l15;           // fixed output column for this lane

    const float bhv = bh[ncol];

    // Preload B fragments (Whh[:, ncol]) once, bf16, kept in registers.
    // B layout (32x16 bf16): elems 0..7 = K(kc*32 + sel*8 + 0..7), elems 8..15 = +16, at col n.
    v16bf Bfrag[8];
#pragma unroll
    for (int kc = 0; kc < 8; ++kc) {
        const int kbase = kc * 32 + sel * 8;
#pragma unroll
        for (int j = 0; j < 16; ++j)
            Bfrag[kc][j] = (__bf16)Whh[(kbase + (j & 7) + (j >> 3) * 16) * HID + ncol];
    }

    // Initial state -> bf16 LDS; row indices for t=0.
    for (int i = tid; i < BATCH * HID; i += 512)
        sH[0][i] = (__bf16)state[i];
    if (tid < BATCH) sRow[0][tid] = inp[tid * TLEN + 0];
    __syncthreads();

    for (int t = 0; t < TLEN; ++t) {
        const int cur = t & 1;
        const int nxt = cur ^ 1;
        const __bf16* hsrc = sH[cur];
        const int*    rows = sRow[cur];

        // stage next step's gather indices into the other buffer
        if (tid < BATCH && (t + 1) < TLEN)
            sRow[nxt][tid] = inp[tid * TLEN + (t + 1)];

#pragma unroll 1
        for (int mt = 0; mt < 8; ++mt) {
            const int mb = mt * 16;

            // prefetch x = Wxh[idx[m], ncol]; 32-bit offsets -> saddr+scale_offset
            float xv[8];
#pragma unroll
            for (int i = 0; i < 8; ++i) {
                const int m = mb + i + sel * 8;
                xv[i] = Wxh[(unsigned)(rows[m] * HID + ncol)];
            }

            // load ALL A fragments (kept in distinct regs by the fence below)
            // A (16x32 bf16): elems 0..7 = K(k0+sel*8+0..7), elems 8..15 = +16
            FragBF A[8];
            const __bf16* arowp = hsrc + (mb + l15) * HID + sel * 8;
#pragma unroll
            for (int kc = 0; kc < 8; ++kc) {
                A[kc].u[0] = *(const uint4*)(arowp + kc * 32);
                A[kc].u[1] = *(const uint4*)(arowp + kc * 32 + 16);
            }
            SCHED_FENCE();

            v8f acc = {};
#pragma unroll
            for (int kc = 0; kc < 8; ++kc)
                acc = __builtin_amdgcn_wmma_f32_16x16x32_bf16(
                          false, A[kc].v, false, Bfrag[kc], (short)0, acc, false, false);
            SCHED_FENCE();

            // epilogue: tanh then stores; h_final branch hoisted (uniform)
            float hnew[8];
#pragma unroll
            for (int i = 0; i < 8; ++i)
                hnew[i] = FAST_TANH(acc[i] + xv[i] + bhv);

#pragma unroll
            for (int i = 0; i < 8; ++i) {
                const int m = mb + i + sel * 8;          // batch row
                const __bf16 hb = (__bf16)hnew[i];
                sH[nxt][m * HID + ncol] = hb;
                Ybf[(unsigned)((t * BATCH + m) * HID + ncol)] = hb;
            }
            if (t == TLEN - 1) {
#pragma unroll
                for (int i = 0; i < 8; ++i) {
                    const int m = mb + i + sel * 8;
                    hfin[m * HID + ncol] = hnew[i];
                }
            }
        }
        __syncthreads();
    }
}

// ---------------------------------------------------------------------------
// Kernel 3: out[65536,1024] = Ybf @ Wd_bf16 + bd   (f32 out, store-bound)
// 128x128 block tile, 8 waves; wave = one 16-row strip across 128 cols.
// ---------------------------------------------------------------------------
__global__ __launch_bounds__(256)
void dense_out(const __bf16* __restrict__ Ybf,  // [TLEN*BATCH, HID]
               const __bf16* __restrict__ Wdb,  // [HID, VOCAB]
               const float*  __restrict__ bd,   // [VOCAB]
               float*        __restrict__ out)  // [TLEN*BATCH, VOCAB]
{
    __shared__ __bf16 sB[128 * HID];  // Wd tile, transposed: sB[n][k], 64 KB

    const int tid  = threadIdx.x;
    const int wave = tid >> 5;
    const int lane = tid & 31;
    const int l15  = lane & 15;
    const int sel  = lane >> 4;

    const int mb0 = blockIdx.x * 128;
    const int nb0 = blockIdx.y * 128;

    // Stage Wd[0:256, nb0:nb0+128] transposed into LDS.
    // Each thread: 4x b128 global loads (4 K-rows x 8 cols), 8x b64 LDS stores.
#pragma unroll
    for (int rep = 0; rep < 4; ++rep) {
        const int c  = rep * 256 + tid;        // 1024 chunks of (4k x 8n)
        const int k0 = (c >> 4) * 4;
        const int n0 = (c & 15) * 8;
        const __bf16* gp = Wdb + (size_t)k0 * VOCAB + nb0 + n0;
        Pack8BF r0, r1, r2, r3;
        r0.u = *(const uint4*)(gp);
        r1.u = *(const uint4*)(gp + VOCAB);
        r2.u = *(const uint4*)(gp + 2 * VOCAB);
        r3.u = *(const uint4*)(gp + 3 * VOCAB);
#pragma unroll
        for (int j = 0; j < 8; ++j) {
            Pack4BF q;
            q.h[0] = r0.h[j]; q.h[1] = r1.h[j];
            q.h[2] = r2.h[j]; q.h[3] = r3.h[j];
            *(uint2*)(sB + (n0 + j) * HID + k0) = q.u;
        }
    }
    __syncthreads();

    const int arow = mb0 + wave * 16 + l15;
    const __bf16* aptr = Ybf + (size_t)arow * HID + sel * 8;

    v8f acc[8];
#pragma unroll
    for (int nt = 0; nt < 8; ++nt) acc[nt] = (v8f){};

#pragma unroll
    for (int kc = 0; kc < 8; ++kc) {
        const int k0 = kc * 32;
        FragBF A;
        A.u[0] = *(const uint4*)(aptr + k0);
        A.u[1] = *(const uint4*)(aptr + k0 + 16);

        // prefetch all 8 B fragments for this k-chunk (distinct regs)
        FragBF B[8];
#pragma unroll
        for (int nt = 0; nt < 8; ++nt) {
            const __bf16* bp = sB + (nt * 16 + l15) * HID + k0 + sel * 8;
            B[nt].u[0] = *(const uint4*)(bp);
            B[nt].u[1] = *(const uint4*)(bp + 16);
        }
        SCHED_FENCE();
#pragma unroll
        for (int nt = 0; nt < 8; ++nt)
            acc[nt] = __builtin_amdgcn_wmma_f32_16x16x32_bf16(
                          false, A.v, false, B[nt].v, (short)0, acc[nt], false, false);
        SCHED_FENCE();
    }

    const int orow0 = mb0 + wave * 16 + sel * 8;
#pragma unroll
    for (int nt = 0; nt < 8; ++nt) {
        const int ncol = nb0 + nt * 16 + l15;
        const float bdv = bd[ncol];
#pragma unroll
        for (int i = 0; i < 8; ++i) {
            const int r = orow0 + i;
            out[(size_t)r * VOCAB + ncol] = acc[nt][i] + bdv;
        }
    }
}

// ---------------------------------------------------------------------------
extern "C" void kernel_launch(void* const* d_in, const int* in_sizes, int n_in,
                              void* d_out, int out_size, void* d_ws, size_t ws_size,
                              hipStream_t stream) {
    const int*   inp   = (const int*)  d_in[0];
    const float* state = (const float*)d_in[1];
    const float* Wxh   = (const float*)d_in[2];
    const float* Whh   = (const float*)d_in[3];
    const float* bh    = (const float*)d_in[4];
    const float* Wd    = (const float*)d_in[5];
    const float* bd    = (const float*)d_in[6];

    float* out = (float*)d_out;

    // workspace: Y bf16 [TLEN*BATCH*HID] then Wd bf16 [HID*VOCAB]
    __bf16* Ybf = (__bf16*)d_ws;
    __bf16* Wdb = (__bf16*)((char*)d_ws + (size_t)TLEN * BATCH * HID * sizeof(__bf16));

    const int n4 = (HID * VOCAB) / 4;
    wd_to_bf16<<<(n4 + 255) / 256, 256, 0, stream>>>((const float4*)Wd, (uint2*)Wdb, n4);

    rnn_scan<<<1, 512, 0, stream>>>(inp, state, Wxh, Whh, bh, Ybf,
                                    out + (size_t)TLEN * BATCH * VOCAB);

    dim3 grid(TLEN * BATCH / 128, VOCAB / 128);
    dense_out<<<grid, 256, 0, stream>>>(Ybf, Wdb, bd, out);
}
